// SupervisedGraphSage_72980084293968
// MI455X (gfx1250) — matrix operand order repaired and checked by reading, not compiled, not run on hardware
//
#include <hip/hip_runtime.h>

// ---------------------------------------------------------------------------
// GraphSAGE 2-layer forward for MI455X (gfx1250, wave32, WMMA bf16)
//
// d_in[0] nodes     int32  [10000]
// d_in[1] neigh_idx int32  [100000,16]
// d_in[2] features  f32    [100000,128]
// d_in[3] W1        f32    [128,256]
// d_in[4] W2        f32    [16,256]
// d_out             f32    [10000,16]
// d_ws:  h1 stored as bf16 [100000,128]  (25.6 MB)
// ---------------------------------------------------------------------------

typedef __attribute__((ext_vector_type(16))) __bf16 v16bf;
typedef __attribute__((ext_vector_type(8)))  __bf16 v8bf;
typedef __attribute__((ext_vector_type(8)))  float  v8f;

#define N_NODES   100000
#define DEG       16
#define N_FEAT    128
#define HID       128
#define N_CLASSES 16
#define BATCH     10000
#define KDIM      256          // 2*N_FEAT == 2*HID
#define LDK       264          // padded K-stride (bf16 elems) -> 528B rows, no LDS bank conflicts
#define L1_TILES  (N_NODES / 16)   // 6250
#define L2_TILES  (BATCH   / 16)   // 625

// Build one v16bf A-fragment (16x32 bf16, ISA 7.12.2 layout):
//   lane = 16*lg + m ; element e<8 -> K = kbase + lg*8 + e
//                      element e>=8 -> K = kbase + 16 + lg*8 + (e-8)
__device__ inline v16bf load_a_frag(const __bf16* row, int kbase, int lg) {
    v8bf lo = *(const v8bf*)(row + kbase + lg * 8);
    v8bf hi = *(const v8bf*)(row + kbase + 16 + lg * 8);
    v16bf a;
#pragma unroll
    for (int e = 0; e < 8; ++e) { a[e] = lo[e]; a[e + 8] = hi[e]; }
    return a;
}

// Build one v16bf B-fragment (32x16 bf16): lane holds column n = l&15,
// element e<8 -> K = kbase + lg*8 + e ; e>=8 -> K = kbase + 16 + lg*8 + (e-8)
__device__ inline v16bf load_b_frag(const __bf16* col_row, int kbase, int lg) {
    v8bf lo = *(const v8bf*)(col_row + kbase + lg * 8);
    v8bf hi = *(const v8bf*)(col_row + kbase + 16 + lg * 8);
    v16bf b;
#pragma unroll
    for (int e = 0; e < 8; ++e) { b[e] = lo[e]; b[e + 8] = hi[e]; }
    return b;
}

// ---------------------------------------------------------------------------
// Layer 1: h1 = relu([features | mean(neigh features)] @ W1^T)   (all nodes)
// Persistent blocks; W1 staged to LDS as bf16 once per block.
// ---------------------------------------------------------------------------
__global__ __launch_bounds__(256) void sage_layer1(
    const int* __restrict__ neigh_idx, const float* __restrict__ features,
    const float* __restrict__ W1, __bf16* __restrict__ h1)
{
    __shared__ __align__(16) __bf16 sW1[HID * LDK];   // 67584 B
    __shared__ __align__(16) __bf16 sA[16 * LDK];     //  8448 B

    const int tid = threadIdx.x;

    // Stage W1 (f32 [128,256]) -> bf16 LDS with padded stride
    for (int idx = tid; idx < HID * KDIM; idx += 256) {
        int n = idx >> 8, k = idx & 255;
        sW1[n * LDK + k] = (__bf16)W1[idx];
    }
    __syncthreads();

    const int g    = tid >> 4;   // node row within tile (0..15)
    const int q    = tid & 15;   // 8-feature chunk id
    const int f0   = q * 8;
    const int lane = tid & 31;
    const int lg   = lane >> 4;  // lane group
    const int m    = lane & 15;  // A row / B,D column
    const int w    = tid >> 5;   // wave id (0..7) -> output N-tile
    const int n0   = w * 16;

    for (int tile = blockIdx.x; tile < L1_TILES; tile += gridDim.x) {
        const int nd = tile * 16 + g;

        // ---- build comb row: [self(128) | neigh-mean(128)] as bf16 ----
        {
            const float4* sv = (const float4*)(features + (size_t)nd * N_FEAT + f0);
            float4 s0 = sv[0], s1 = sv[1];
            __bf16* ds = sA + g * LDK + f0;
            ds[0] = (__bf16)s0.x; ds[1] = (__bf16)s0.y;
            ds[2] = (__bf16)s0.z; ds[3] = (__bf16)s0.w;
            ds[4] = (__bf16)s1.x; ds[5] = (__bf16)s1.y;
            ds[6] = (__bf16)s1.z; ds[7] = (__bf16)s1.w;

            float acc[8];
#pragma unroll
            for (int j = 0; j < 8; ++j) acc[j] = 0.f;
            const int* nb = neigh_idx + (size_t)nd * DEG;
#pragma unroll 4
            for (int d = 0; d < DEG; ++d) {
                const float* nrow = features + (size_t)nb[d] * N_FEAT + f0;
                if (d + 1 < DEG)
                    __builtin_prefetch(features + (size_t)nb[d + 1] * N_FEAT + f0, 0, 0);
                const float4* nv = (const float4*)nrow;
                float4 a0 = nv[0], a1 = nv[1];
                acc[0] += a0.x; acc[1] += a0.y; acc[2] += a0.z; acc[3] += a0.w;
                acc[4] += a1.x; acc[5] += a1.y; acc[6] += a1.z; acc[7] += a1.w;
            }
            __bf16* dn = sA + g * LDK + N_FEAT + f0;
#pragma unroll
            for (int j = 0; j < 8; ++j) dn[j] = (__bf16)(acc[j] * (1.0f / DEG));
        }
        __syncthreads();

        // ---- WMMA: wave w computes [16 x 16] tile at columns n0..n0+15 ----
        {
            const __bf16* arow = sA + m * LDK;          // A row m
            const __bf16* brow = sW1 + (n0 + m) * LDK;  // B column (n0+m) = W1 row
            v8f c = {};
#pragma unroll
            for (int kb = 0; kb < KDIM / 32; ++kb) {
                v16bf a = load_a_frag(arow, kb * 32, lg);
                v16bf b = load_b_frag(brow, kb * 32, lg);
                c = __builtin_amdgcn_wmma_f32_16x16x32_bf16(
                        false, a, false, b, (short)0, c, false, false);
            }
            // D element r -> row (8*lg + r), col (n0 + m); relu; store bf16
#pragma unroll
            for (int r = 0; r < 8; ++r) {
                float v = c[r] > 0.f ? c[r] : 0.f;
                h1[(size_t)(tile * 16 + 8 * lg + r) * HID + n0 + m] = (__bf16)v;
            }
        }
        __syncthreads();   // protect sA before next tile overwrites it
    }
}

// ---------------------------------------------------------------------------
// Layer 2: out = [h1[nodes] | mean(h1[neigh(nodes)])] @ W2^T   (batch only)
// One block per 16-node tile; wave 0 does the single 16x16 WMMA chain.
// ---------------------------------------------------------------------------
__global__ __launch_bounds__(256) void sage_layer2(
    const int* __restrict__ nodes, const int* __restrict__ neigh_idx,
    const __bf16* __restrict__ h1, const float* __restrict__ W2,
    float* __restrict__ out)
{
    __shared__ __align__(16) __bf16 sW2[N_CLASSES * LDK];  // 8448 B
    __shared__ __align__(16) __bf16 sA[16 * LDK];          // 8448 B

    const int tid  = threadIdx.x;
    const int tile = blockIdx.x;

    // Stage W2 (f32 [16,256]) -> bf16 LDS
    for (int idx = tid; idx < N_CLASSES * KDIM; idx += 256) {
        int n = idx >> 8, k = idx & 255;
        sW2[n * LDK + k] = (__bf16)W2[idx];
    }

    // ---- build comb2 tile ----
    {
        const int g  = tid >> 4;
        const int f0 = (tid & 15) * 8;
        const int b  = tile * 16 + g;
        const int nd = nodes[b];

        // self (already bf16): straight copy
        v8bf hs = *(const v8bf*)(h1 + (size_t)nd * HID + f0);
        *(v8bf*)(sA + g * LDK + f0) = hs;

        // neighbor mean in f32
        float acc[8];
#pragma unroll
        for (int j = 0; j < 8; ++j) acc[j] = 0.f;
        const int* nb = neigh_idx + (size_t)nd * DEG;
#pragma unroll 4
        for (int d = 0; d < DEG; ++d) {
            const __bf16* hrow = h1 + (size_t)nb[d] * HID + f0;
            if (d + 1 < DEG)
                __builtin_prefetch(h1 + (size_t)nb[d + 1] * HID + f0, 0, 0);
            v8bf hv = *(const v8bf*)hrow;
#pragma unroll
            for (int j = 0; j < 8; ++j) acc[j] += (float)hv[j];
        }
        __bf16* dn = sA + g * LDK + HID + f0;
#pragma unroll
        for (int j = 0; j < 8; ++j) dn[j] = (__bf16)(acc[j] * (1.0f / DEG));
    }
    __syncthreads();

    // ---- wave 0: 16x16 output tile, K = 256 -> 8 bf16 WMMAs ----
    if (tid < 32) {
        const int lane = tid;
        const int lg   = lane >> 4;
        const int m    = lane & 15;
        const __bf16* arow = sA + m * LDK;
        const __bf16* brow = sW2 + m * LDK;   // n0 = 0, N == 16 classes
        v8f c = {};
#pragma unroll
        for (int kb = 0; kb < KDIM / 32; ++kb) {
            v16bf a = load_a_frag(arow, kb * 32, lg);
            v16bf b = load_b_frag(brow, kb * 32, lg);
            c = __builtin_amdgcn_wmma_f32_16x16x32_bf16(
                    false, a, false, b, (short)0, c, false, false);
        }
#pragma unroll
        for (int r = 0; r < 8; ++r)
            out[(size_t)(tile * 16 + 8 * lg + r) * N_CLASSES + m] = c[r];
    }
}

// ---------------------------------------------------------------------------
extern "C" void kernel_launch(void* const* d_in, const int* in_sizes, int n_in,
                              void* d_out, int out_size, void* d_ws, size_t ws_size,
                              hipStream_t stream) {
    const int*   nodes     = (const int*)  d_in[0];
    const int*   neigh_idx = (const int*)  d_in[1];
    const float* features  = (const float*)d_in[2];
    const float* W1        = (const float*)d_in[3];
    const float* W2        = (const float*)d_in[4];
    float*       out       = (float*)d_out;
    __bf16*      h1        = (__bf16*)d_ws;   // 100000*128*2 B = 25.6 MB

    sage_layer1<<<1024, 256, 0, stream>>>(neigh_idx, features, W1, h1);
    sage_layer2<<<L2_TILES, 256, 0, stream>>>(nodes, neigh_idx, h1, W2, out);
}
